// SComGNN_40218073759806
// MI455X (gfx1250) — compile-verified, matrix operand order checked by reading, not run on hardware
//
#include <hip/hip_runtime.h>
#include <hip/hip_bf16.h>

#define NN 50000
#define EE 1600000
#define DD 128
#define CATC 768
#define BB 4096
#define NNEG4 4

typedef __attribute__((ext_vector_type(16))) __bf16 v16bf;
typedef __attribute__((ext_vector_type(8)))  float  v8f;

__device__ inline unsigned short f2bf16u(float f) {
    unsigned u = __builtin_bit_cast(unsigned, f);
    unsigned r = u + 0x7FFFu + ((u >> 16) & 1u);
    return (unsigned short)(r >> 16);
}

// ---------------- small fp32 GEMM: C[M x 128] = A[M x K] @ B[K x 128] ----------------
__global__ void small_gemm_f32(const float* __restrict__ A, const float* __restrict__ B,
                               float* __restrict__ C, int M, int K) {
    int idx = blockIdx.x * blockDim.x + threadIdx.x;
    if (idx >= M * 128) return;
    int m = idx >> 7, n = idx & 127;
    float s = 0.f;
    for (int k = 0; k < K; ++k) s += A[(size_t)m * K + k] * B[(size_t)k * 128 + n];
    C[idx] = s;
}

// bias_f[d] = b_emb[d] + sum_k b_cid2[k]*Wemb[k][d] + b_cid3[k]*Wemb[128+k][d]
__global__ void fuse_bias(const float* b2, const float* b3, const float* bemb,
                          const float* Wemb, float* out) {
    int d = threadIdx.x;
    float s = bemb[d];
    for (int k = 0; k < 128; ++k)
        s += b2[k] * Wemb[(size_t)k * 128 + d] + b3[k] * Wemb[(size_t)(128 + k) * 128 + d];
    out[d] = s;
}

// ---------------- weight swizzle into WMMA B-fragment order ----------------
// frag index: ((kt*8 + nt)*32 + lane)*16 + e ; B value = W[kt*32 + kmap(lane,e)][nt*16 + (lane&15)]
__global__ void swizzle_w(const float* __restrict__ W, unsigned short* __restrict__ out, int K) {
    int idx = blockIdx.x * blockDim.x + threadIdx.x;
    int total = (K >> 5) * 8 * 32;
    if (idx >= total) return;
    int lane = idx & 31;
    int nt = (idx >> 5) & 7;
    int kt = idx >> 8;
    int col = nt * 16 + (lane & 15);
    int kbase = kt * 32 + ((lane >> 4) << 3);
    unsigned short* dst = out + (size_t)idx * 16;
#pragma unroll
    for (int e = 0; e < 16; ++e) {
        int k = kbase + (e < 8 ? e : e + 8);
        dst[e] = f2bf16u(W[(size_t)k * 128 + col]);
    }
}

// ---------------- WMMA GEMM main loop (single basic block, in-place accumulation) ----
template <bool HASY>
__device__ __forceinline__ void gemm_loop(v8f acc[8], const float* __restrict__ xrow,
                                          const float* __restrict__ yrow,
                                          float alpha, float beta,
                                          const v16bf* __restrict__ Wfrag,
                                          int ktiles, int lane) {
    // prologue: load A fragment for kt=0
    float a[16];
#pragma unroll
    for (int e = 0; e < 8; ++e) a[e] = xrow[e];
#pragma unroll
    for (int e = 0; e < 8; ++e) a[8 + e] = xrow[16 + e];
    if (HASY) {
#pragma unroll
        for (int e = 0; e < 8; ++e) a[e] = alpha * a[e] + beta * yrow[e];
#pragma unroll
        for (int e = 0; e < 8; ++e) a[8 + e] = alpha * a[8 + e] + beta * yrow[16 + e];
    }

    for (int kt = 0; kt < ktiles; ++kt) {
        // preload all 8 B fragments for this K-tile
        const v16bf* wrow = Wfrag + (size_t)kt * 256 + lane;
        v16bf bv0 = wrow[0 * 32];
        v16bf bv1 = wrow[1 * 32];
        v16bf bv2 = wrow[2 * 32];
        v16bf bv3 = wrow[3 * 32];
        v16bf bv4 = wrow[4 * 32];
        v16bf bv5 = wrow[5 * 32];
        v16bf bv6 = wrow[6 * 32];
        v16bf bv7 = wrow[7 * 32];

        // convert current A fragment to bf16, then overwrite `a` with next tile
        v16bf av;
#pragma unroll
        for (int e = 0; e < 16; ++e) av[e] = (__bf16)a[e];

        int ktn = kt + ((kt + 1 < ktiles) ? 1 : 0);  // branchless clamp
        const float* xn = xrow + (size_t)ktn * 32;
        __builtin_prefetch(xn + 64, 0, 1);
#pragma unroll
        for (int e = 0; e < 8; ++e) a[e] = xn[e];
#pragma unroll
        for (int e = 0; e < 8; ++e) a[8 + e] = xn[16 + e];
        if (HASY) {
            const float* yn = yrow + (size_t)ktn * 32;
#pragma unroll
            for (int e = 0; e < 8; ++e) a[e] = alpha * a[e] + beta * yn[e];
#pragma unroll
            for (int e = 0; e < 8; ++e) a[8 + e] = alpha * a[8 + e] + beta * yn[16 + e];
        }

        acc[0] = __builtin_amdgcn_wmma_f32_16x16x32_bf16(false, av, false, bv0, (short)0, acc[0], false, false);
        acc[1] = __builtin_amdgcn_wmma_f32_16x16x32_bf16(false, av, false, bv1, (short)0, acc[1], false, false);
        acc[2] = __builtin_amdgcn_wmma_f32_16x16x32_bf16(false, av, false, bv2, (short)0, acc[2], false, false);
        acc[3] = __builtin_amdgcn_wmma_f32_16x16x32_bf16(false, av, false, bv3, (short)0, acc[3], false, false);
        acc[4] = __builtin_amdgcn_wmma_f32_16x16x32_bf16(false, av, false, bv4, (short)0, acc[4], false, false);
        acc[5] = __builtin_amdgcn_wmma_f32_16x16x32_bf16(false, av, false, bv5, (short)0, acc[5], false, false);
        acc[6] = __builtin_amdgcn_wmma_f32_16x16x32_bf16(false, av, false, bv6, (short)0, acc[6], false, false);
        acc[7] = __builtin_amdgcn_wmma_f32_16x16x32_bf16(false, av, false, bv7, (short)0, acc[7], false, false);
    }
}

// ---------------- main WMMA GEMM: out[M x 128] = act(Aeff[M x K] @ Wswz + bias + extra) ----
// Aeff = (Y ? alpha*X + beta*Y : X)
__global__ __launch_bounds__(256) void gemm_nk128(
    int M, int K,
    const float* __restrict__ X, const float* __restrict__ Y, float alpha, float beta,
    const unsigned short* __restrict__ Wswz, const float* __restrict__ bias,
    const float* __restrict__ extra_table, const int* __restrict__ extra_idx,
    int relu, float* __restrict__ out) {
    int wave = threadIdx.x >> 5;
    int lane = threadIdx.x & 31;
    int rowBase = blockIdx.x * 128 + wave * 16;
    int r = rowBase + (lane & 15);
    int rc = (r < M) ? r : (M - 1);
    int khalf = (lane >> 4) << 3;

    v8f acc[8];
#pragma unroll
    for (int i = 0; i < 8; ++i)
#pragma unroll
        for (int e = 0; e < 8; ++e) acc[i][e] = 0.f;

    const v16bf* Wfrag = (const v16bf*)Wswz;
    int ktiles = K >> 5;
    const float* xrow = X + (size_t)rc * K + khalf;

    if (Y) {
        const float* yrow = Y + (size_t)rc * K + khalf;
        gemm_loop<true>(acc, xrow, yrow, alpha, beta, Wfrag, ktiles, lane);
    } else {
        gemm_loop<false>(acc, xrow, nullptr, alpha, beta, Wfrag, ktiles, lane);
    }

    int colb = lane & 15;
    int rowOff = (lane >= 16) ? 8 : 0;
#pragma unroll
    for (int nt = 0; nt < 8; ++nt) {
        int col = nt * 16 + colb;
        float bse = bias ? bias[col] : 0.f;
#pragma unroll
        for (int e = 0; e < 8; ++e) {
            int row = rowBase + rowOff + e;
            if (row < M) {
                float v = acc[nt][e] + bse;
                if (extra_table) v += extra_table[(size_t)extra_idx[row] * 128 + col];
                if (relu) v = fmaxf(v, 0.f);
                out[(size_t)row * 128 + col] = v;
            }
        }
    }
}

// ---------------- SpMM via float atomics: y[row] += val * x[col], one edge per wave ----
__global__ __launch_bounds__(256) void spmm_atomic(
    const int* __restrict__ row, const int* __restrict__ col, const float* __restrict__ val,
    const float* __restrict__ x, float* __restrict__ y, int E) {
    int e = blockIdx.x * 8 + (threadIdx.x >> 5);
    int lane = threadIdx.x & 31;
    if (e >= E) return;
    int r = row[e], c = col[e];
    float v = val[e];
    float4 d = ((const float4*)(x + (size_t)c * 128))[lane];
    float* yp = y + (size_t)r * 128 + lane * 4;
    atomicAdd(yp + 0, v * d.x);
    atomicAdd(yp + 1, v * d.y);
    atomicAdd(yp + 2, v * d.z);
    atomicAdd(yp + 3, v * d.w);
}

// ---------------- column stats ----------------
__global__ void col_stats(const float* __restrict__ x, int M,
                          float* __restrict__ sum, float* __restrict__ sumsq) {
    int colc = threadIdx.x & 127;
    int part = blockIdx.x * 2 + (threadIdx.x >> 7);
    int nparts = gridDim.x * 2;
    float s = 0.f, q = 0.f;
    for (int r = part; r < M; r += nparts) {
        float v = x[(size_t)r * 128 + colc];
        s += v;
        q += v * v;
    }
    atomicAdd(&sum[colc], s);
    atomicAdd(&sumsq[colc], q);
}

__global__ void bn_finalize(const float* sum, const float* sq, const float* g, const float* b,
                            float invM, float* scale, float* shift) {
    int d = threadIdx.x;
    float m = sum[d] * invM;
    float var = sq[d] * invM - m * m;
    float inv = rsqrtf(var + 1e-5f);
    float sc = g[d] * inv;
    scale[d] = sc;
    shift[d] = b[d] - m * sc;
}

// ---------------- gather + normalize into key/pos/neg ----------------
__global__ void gather_bn(const int* __restrict__ train,
                          const float* __restrict__ low, const float* __restrict__ mid,
                          const float* sc1, const float* sh1, const float* sc2, const float* sh2,
                          float* __restrict__ key_e, float* __restrict__ pos_e,
                          float* __restrict__ neg_e) {
    int t = blockIdx.x * blockDim.x + threadIdx.x;
    if (t >= BB * 6 * 128) return;
    int d = t & 127;
    int slot = t >> 7;
    int j = slot % 6, b = slot / 6;
    int n = train[b * 6 + j];
    float lv = low[(size_t)n * 128 + d] * sc1[d] + sh1[d];
    float mv = mid[(size_t)n * 128 + d] * sc2[d] + sh2[d];
    float* dst;
    if (j == 0) dst = key_e + (size_t)b * 256;
    else if (j == 1) dst = pos_e + (size_t)b * 256;
    else dst = neg_e + (size_t)(b * 4 + (j - 2)) * 256;
    dst[d] = lv;
    dst[128 + d] = mv;
}

__global__ void expand_rep(const float* __restrict__ key_e, float* __restrict__ key_rep) {
    int t = blockIdx.x * blockDim.x + threadIdx.x;
    if (t >= BB * NNEG4 * 256) return;
    int d = t & 255;
    int i = t >> 8;
    key_rep[(size_t)i * 256 + d] = key_e[(size_t)(i >> 2) * 256 + d];
}

// ---------------- 2x2 attention core (softmax over q axis!) ----------------
__global__ __launch_bounds__(256) void att_core(
    const float* __restrict__ q, const float* __restrict__ k, const float* __restrict__ v,
    float* __restrict__ h, int Bn) {
    int b = blockIdx.x * 8 + (threadIdx.x >> 5);
    int lane = threadIdx.x & 31;
    if (b >= Bn) return;
    float4 q0 = ((const float4*)(q + (size_t)b * 256))[lane];
    float4 q1 = ((const float4*)(q + (size_t)b * 256 + 128))[lane];
    float4 k0 = ((const float4*)(k + (size_t)b * 256))[lane];
    float4 k1 = ((const float4*)(k + (size_t)b * 256 + 128))[lane];
    float4 v0 = ((const float4*)(v + (size_t)b * 256))[lane];
    float4 v1 = ((const float4*)(v + (size_t)b * 256 + 128))[lane];
    float s00 = q0.x * k0.x + q0.y * k0.y + q0.z * k0.z + q0.w * k0.w;
    float s01 = q0.x * k1.x + q0.y * k1.y + q0.z * k1.z + q0.w * k1.w;
    float s10 = q1.x * k0.x + q1.y * k0.y + q1.z * k0.z + q1.w * k0.w;
    float s11 = q1.x * k1.x + q1.y * k1.y + q1.z * k1.z + q1.w * k1.w;
#pragma unroll
    for (int off = 16; off > 0; off >>= 1) {
        s00 += __shfl_xor(s00, off, 32);
        s01 += __shfl_xor(s01, off, 32);
        s10 += __shfl_xor(s10, off, 32);
        s11 += __shfl_xor(s11, off, 32);
    }
    const float scl = 0.08838834764831845f; // 1/sqrt(128)
    s00 *= scl; s01 *= scl; s10 *= scl; s11 *= scl;
    // softmax over axis=1 (q index) per k column
    float m0 = fmaxf(s00, s10), m1 = fmaxf(s01, s11);
    float e00 = __expf(s00 - m0), e10 = __expf(s10 - m0);
    float e01 = __expf(s01 - m1), e11 = __expf(s11 - m1);
    float r0 = 1.f / (e00 + e10), r1 = 1.f / (e01 + e11);
    float a00 = e00 * r0, a10 = e10 * r0, a01 = e01 * r1, a11 = e11 * r1;
    float4 h0, h1;
    h0.x = a00 * v0.x + a01 * v1.x; h0.y = a00 * v0.y + a01 * v1.y;
    h0.z = a00 * v0.z + a01 * v1.z; h0.w = a00 * v0.w + a01 * v1.w;
    h1.x = a10 * v0.x + a11 * v1.x; h1.y = a10 * v0.y + a11 * v1.y;
    h1.z = a10 * v0.z + a11 * v1.z; h1.w = a10 * v0.w + a11 * v1.w;
    ((float4*)(h + (size_t)b * 256))[lane] = h0;
    ((float4*)(h + (size_t)b * 256 + 128))[lane] = h1;
}

// ---------------- loss ----------------
__global__ __launch_bounds__(256) void loss_kernel(
    const float* __restrict__ klp, const float* __restrict__ pl,
    const float* __restrict__ kln, const float* __restrict__ nl, float* accum) {
    int w = blockIdx.x * 8 + (threadIdx.x >> 5);
    int lane = threadIdx.x & 31;
    if (w >= BB * NNEG4) return;
    int b = w >> 2;
    float4 kp = ((const float4*)(klp + (size_t)b * 128))[lane];
    float4 pp = ((const float4*)(pl + (size_t)b * 128))[lane];
    float4 kn = ((const float4*)(kln + (size_t)w * 128))[lane];
    float4 nn = ((const float4*)(nl + (size_t)w * 128))[lane];
    float ps = kp.x * pp.x + kp.y * pp.y + kp.z * pp.z + kp.w * pp.w;
    float ns = kn.x * nn.x + kn.y * nn.y + kn.z * nn.z + kn.w * nn.w;
#pragma unroll
    for (int off = 16; off > 0; off >>= 1) {
        ps += __shfl_xor(ps, off, 32);
        ns += __shfl_xor(ns, off, 32);
    }
    if (lane == 0) {
        float x = ps - ns;
        float sig = 1.f / (1.f + __expf(-x));
        atomicAdd(accum, __logf(sig + 1e-9f));
    }
}

__global__ void loss_finalize(const float* accum, float* out) {
    out[0] = -accum[0] / (float)(BB * NNEG4);
}

// ---------------- host-side helpers ----------------
struct AttW {
    const unsigned short *q1, *k1, *v1, *q2, *k2, *v2, *cat;
    const float *bq1, *bk1, *bv1, *bq2, *bk2, *bv2, *bcat;
};

static void run_two_att(int Bn, const float* qx, const float* x, float* outp, const AttW& w,
                        float* qb, float* kb, float* vb, float* hb, hipStream_t stream) {
    int M = Bn * 2;
    int gb = (M + 127) / 128;
    gemm_nk128<<<gb, 256, 0, stream>>>(M, 128, qx, nullptr, 1.f, 0.f, w.q1, w.bq1, nullptr, nullptr, 0, qb);
    gemm_nk128<<<gb, 256, 0, stream>>>(M, 128, x,  nullptr, 1.f, 0.f, w.k1, w.bk1, nullptr, nullptr, 0, kb);
    gemm_nk128<<<gb, 256, 0, stream>>>(M, 128, x,  nullptr, 1.f, 0.f, w.v1, w.bv1, nullptr, nullptr, 0, vb);
    att_core<<<(Bn + 7) / 8, 256, 0, stream>>>(qb, kb, vb, hb, Bn);
    gemm_nk128<<<gb, 256, 0, stream>>>(M, 128, hb, nullptr, 1.f, 0.f, w.q2, w.bq2, nullptr, nullptr, 0, qb);
    gemm_nk128<<<gb, 256, 0, stream>>>(M, 128, hb, nullptr, 1.f, 0.f, w.k2, w.bk2, nullptr, nullptr, 0, kb);
    gemm_nk128<<<gb, 256, 0, stream>>>(M, 128, hb, nullptr, 1.f, 0.f, w.v2, w.bv2, nullptr, nullptr, 0, vb);
    att_core<<<(Bn + 7) / 8, 256, 0, stream>>>(qb, kb, vb, hb, Bn);
    gemm_nk128<<<(Bn + 127) / 128, 256, 0, stream>>>(Bn, 256, hb, nullptr, 1.f, 0.f, w.cat, w.bcat, nullptr, nullptr, 0, outp);
}

extern "C" void kernel_launch(void* const* d_in, const int* in_sizes, int n_in,
                              void* d_out, int out_size, void* d_ws, size_t ws_size,
                              hipStream_t stream) {
    const float* features = (const float*)d_in[0];
    const int*   price    = (const int*)d_in[1];
    const int*   adj_row  = (const int*)d_in[2];
    const int*   adj_col  = (const int*)d_in[3];
    const float* adj_val  = (const float*)d_in[4];
    const int*   train    = (const int*)d_in[5];
    const float* W_cid2 = (const float*)d_in[6];
    const float* b_cid2 = (const float*)d_in[7];
    const float* W_cid3 = (const float*)d_in[8];
    const float* b_cid3 = (const float*)d_in[9];
    const float* price_table = (const float*)d_in[10];
    const float* W_emb = (const float*)d_in[11];
    const float* b_emb = (const float*)d_in[12];
    const float* W_low = (const float*)d_in[13];
    const float* W_mid = (const float*)d_in[14];
    const float* bn1_g = (const float*)d_in[15];
    const float* bn1_b = (const float*)d_in[16];
    const float* bn2_g = (const float*)d_in[17];
    const float* bn2_b = (const float*)d_in[18];
    const float* Wq1 = (const float*)d_in[19]; const float* bq1 = (const float*)d_in[20];
    const float* Wk1 = (const float*)d_in[21]; const float* bk1 = (const float*)d_in[22];
    const float* Wv1 = (const float*)d_in[23]; const float* bv1 = (const float*)d_in[24];
    const float* Wq2 = (const float*)d_in[25]; const float* bq2 = (const float*)d_in[26];
    const float* Wk2 = (const float*)d_in[27]; const float* bk2 = (const float*)d_in[28];
    const float* Wv2 = (const float*)d_in[29]; const float* bv2 = (const float*)d_in[30];
    const float* Wcat = (const float*)d_in[31]; const float* bcat = (const float*)d_in[32];

    char* ws = (char*)d_ws;
    size_t off = 0;
    auto alloc = [&](size_t bytes) -> void* {
        void* p = ws + off;
        off = (off + bytes + 255) & ~(size_t)255;
        return p;
    };

    float* Wbig = (float*)alloc((size_t)1536 * 128 * 4);
    float* price_proj = (float*)alloc((size_t)100 * 128 * 4);
    float* bias_f = (float*)alloc(128 * 4);
    unsigned short* WbigS = (unsigned short*)alloc((size_t)1536 * 128 * 2);
    unsigned short* WlowS = (unsigned short*)alloc((size_t)128 * 128 * 2);
    unsigned short* WmidS = (unsigned short*)alloc((size_t)128 * 128 * 2);
    unsigned short* Wq1S = (unsigned short*)alloc((size_t)128 * 128 * 2);
    unsigned short* Wk1S = (unsigned short*)alloc((size_t)128 * 128 * 2);
    unsigned short* Wv1S = (unsigned short*)alloc((size_t)128 * 128 * 2);
    unsigned short* Wq2S = (unsigned short*)alloc((size_t)128 * 128 * 2);
    unsigned short* Wk2S = (unsigned short*)alloc((size_t)128 * 128 * 2);
    unsigned short* Wv2S = (unsigned short*)alloc((size_t)128 * 128 * 2);
    unsigned short* WcatS = (unsigned short*)alloc((size_t)256 * 128 * 2);
    float* stats = (float*)alloc(4 * 128 * 4);  // sumL, sqL, sumM, sqM
    float* bnp = (float*)alloc(4 * 128 * 4);    // sc1, sh1, sc2, sh2
    float* item = (float*)alloc((size_t)NN * 128 * 4);
    float* ag   = (float*)alloc((size_t)NN * 128 * 4);
    float* ag2  = (float*)alloc((size_t)NN * 128 * 4);
    float* low  = (float*)alloc((size_t)NN * 128 * 4);
    float* mid  = (float*)alloc((size_t)NN * 128 * 4);
    float* key_e = (float*)alloc((size_t)BB * 256 * 4);
    float* pos_e = (float*)alloc((size_t)BB * 256 * 4);
    float* neg_e = (float*)alloc((size_t)BB * NNEG4 * 256 * 4);
    float* key_rep = (float*)alloc((size_t)BB * NNEG4 * 256 * 4);
    float* qb = (float*)alloc((size_t)BB * NNEG4 * 2 * 128 * 4);
    float* kb = (float*)alloc((size_t)BB * NNEG4 * 2 * 128 * 4);
    float* vb = (float*)alloc((size_t)BB * NNEG4 * 2 * 128 * 4);
    float* hb = (float*)alloc((size_t)BB * NNEG4 * 2 * 128 * 4);
    float* klp = (float*)alloc((size_t)BB * 128 * 4);
    float* pl  = (float*)alloc((size_t)BB * 128 * 4);
    float* kln = (float*)alloc((size_t)BB * NNEG4 * 128 * 4);
    float* nl  = (float*)alloc((size_t)BB * NNEG4 * 128 * 4);
    float* lossacc = (float*)alloc(256);

    // ---- weight prep (tiny) ----
    small_gemm_f32<<<(768 * 128 + 255) / 256, 256, 0, stream>>>(W_cid2, W_emb, Wbig, 768, 128);
    small_gemm_f32<<<(768 * 128 + 255) / 256, 256, 0, stream>>>(W_cid3, W_emb + 128 * 128, Wbig + (size_t)768 * 128, 768, 128);
    small_gemm_f32<<<(100 * 128 + 255) / 256, 256, 0, stream>>>(price_table, W_emb + 256 * 128, price_proj, 100, 128);
    fuse_bias<<<1, 128, 0, stream>>>(b_cid2, b_cid3, b_emb, W_emb, bias_f);
    swizzle_w<<<48, 256, 0, stream>>>(Wbig, WbigS, 1536);
    swizzle_w<<<4, 256, 0, stream>>>(W_low, WlowS, 128);
    swizzle_w<<<4, 256, 0, stream>>>(W_mid, WmidS, 128);
    swizzle_w<<<4, 256, 0, stream>>>(Wq1, Wq1S, 128);
    swizzle_w<<<4, 256, 0, stream>>>(Wk1, Wk1S, 128);
    swizzle_w<<<4, 256, 0, stream>>>(Wv1, Wv1S, 128);
    swizzle_w<<<4, 256, 0, stream>>>(Wq2, Wq2S, 128);
    swizzle_w<<<4, 256, 0, stream>>>(Wk2, Wk2S, 128);
    swizzle_w<<<4, 256, 0, stream>>>(Wv2, Wv2S, 128);
    swizzle_w<<<8, 256, 0, stream>>>(Wcat, WcatS, 256);

    // ---- item = relu(features @ Wbig + price_proj[price] + bias_f) ----
    gemm_nk128<<<(NN + 127) / 128, 256, 0, stream>>>(NN, 1536, features, nullptr, 1.f, 0.f,
                                                     WbigS, bias_f, price_proj, price, 1, item);

    // ---- SpMM x2 ----
    hipMemsetAsync(ag, 0, (size_t)NN * 128 * 4, stream);
    spmm_atomic<<<(EE + 7) / 8, 256, 0, stream>>>(adj_row, adj_col, adj_val, item, ag, EE);
    hipMemsetAsync(ag2, 0, (size_t)NN * 128 * 4, stream);
    spmm_atomic<<<(EE + 7) / 8, 256, 0, stream>>>(adj_row, adj_col, adj_val, ag, ag2, EE);

    // ---- low / mid ----
    gemm_nk128<<<(NN + 127) / 128, 256, 0, stream>>>(NN, 128, ag, item, 0.5f, 0.5f,
                                                     WlowS, nullptr, nullptr, nullptr, 0, low);
    gemm_nk128<<<(NN + 127) / 128, 256, 0, stream>>>(NN, 128, ag2, item, 0.5f, -0.5f,
                                                     WmidS, nullptr, nullptr, nullptr, 0, mid);

    // ---- batchnorm stats + params ----
    hipMemsetAsync(stats, 0, 4 * 128 * 4, stream);
    col_stats<<<256, 256, 0, stream>>>(low, NN, stats + 0, stats + 128);
    col_stats<<<256, 256, 0, stream>>>(mid, NN, stats + 256, stats + 384);
    float invM = 1.0f / (float)NN;
    bn_finalize<<<1, 128, 0, stream>>>(stats + 0, stats + 128, bn1_g, bn1_b, invM, bnp + 0, bnp + 128);
    bn_finalize<<<1, 128, 0, stream>>>(stats + 256, stats + 384, bn2_g, bn2_b, invM, bnp + 256, bnp + 384);

    // ---- gather train_set rows, normalized ----
    gather_bn<<<(BB * 6 * 128 + 255) / 256, 256, 0, stream>>>(train, low, mid,
        bnp + 0, bnp + 128, bnp + 256, bnp + 384, key_e, pos_e, neg_e);
    expand_rep<<<(BB * NNEG4 * 256 + 255) / 256, 256, 0, stream>>>(key_e, key_rep);

    // ---- four attention passes ----
    AttW w{Wq1S, Wk1S, Wv1S, Wq2S, Wk2S, Wv2S, WcatS,
           bq1, bk1, bv1, bq2, bk2, bv2, bcat};
    run_two_att(BB, pos_e, key_e, klp, w, qb, kb, vb, hb, stream);
    run_two_att(BB, key_e, pos_e, pl, w, qb, kb, vb, hb, stream);
    run_two_att(BB * NNEG4, neg_e, key_rep, kln, w, qb, kb, vb, hb, stream);
    run_two_att(BB * NNEG4, key_rep, neg_e, nl, w, qb, kb, vb, hb, stream);

    // ---- loss ----
    hipMemsetAsync(lossacc, 0, 4, stream);
    loss_kernel<<<(BB * NNEG4 + 7) / 8, 256, 0, stream>>>(klp, pl, kln, nl, lossacc);
    loss_finalize<<<1, 1, 0, stream>>>(lossacc, (float*)d_out);

    (void)in_sizes; (void)n_in; (void)out_size; (void)ws_size;
}
